// Linear8bitLt_65738769432993
// MI455X (gfx1250) — 8-bit matrix operand order repaired and checked by reading, compiled, not run on hardware
//
#include <hip/hip_runtime.h>
#include <stdint.h>

#define D_IN   4096
#define D_OUT  16384
#define NTOK   4096
#define THRESH 6.0f
#define NC     32

typedef int v8i __attribute__((ext_vector_type(8)));

__global__ void k_init(unsigned int* colmax, int* cnt) {
    int i = blockIdx.x * 256 + threadIdx.x;
    if (i < D_IN) colmax[i] = 0u;
    if (i == 0) *cnt = 0;
}

__global__ void k_colmax(const float* __restrict__ x, unsigned int* colmax) {
    int col = blockIdx.x * 256 + threadIdx.x;
    float m = 0.0f;
    for (int r = blockIdx.y; r < NTOK; r += gridDim.y)
        m = fmaxf(m, fabsf(x[(size_t)r * D_IN + col]));
    atomicMax(&colmax[col], __float_as_uint(m));
}

__global__ void k_build_list(const unsigned int* __restrict__ colmax,
                             int* __restrict__ cnt, int* __restrict__ list) {
    if (threadIdx.x == 0 && blockIdx.x == 0) {
        int n = 0;
        for (int c = 0; c < D_IN; ++c)
            if (__uint_as_float(colmax[c]) > THRESH) list[n++] = c;
        *cnt = n;
    }
}

__global__ void k_quant_x(const float* __restrict__ src, int8_t* __restrict__ dst,
                          float* __restrict__ scale,
                          const unsigned int* __restrict__ colmax) {
    const int row = blockIdx.x;
    const int t   = threadIdx.x;
    __shared__ float red[256];
    float vals[16];
    float m = 0.0f;
    #pragma unroll
    for (int i = 0; i < 16; ++i) {
        int c = t + i * 256;
        float v = src[(size_t)row * D_IN + c];
        if (__uint_as_float(colmax[c]) > THRESH) v = 0.0f;
        vals[i] = v;
        m = fmaxf(m, fabsf(v));
    }
    red[t] = m;
    __syncthreads();
    for (int s = 128; s > 0; s >>= 1) {
        if (t < s) red[t] = fmaxf(red[t], red[t + s]);
        __syncthreads();
    }
    float absmax = fmaxf(red[0], 1e-8f);
    float sc = 127.0f / absmax;
    #pragma unroll
    for (int i = 0; i < 16; ++i) {
        int c = t + i * 256;
        float q = fminf(fmaxf(rintf(vals[i] * sc), -127.0f), 127.0f);
        dst[(size_t)row * D_IN + c] = (int8_t)q;
    }
    if (t == 0) scale[row] = absmax;
}

__global__ void k_rowabsmax_W(const float* __restrict__ src, float* __restrict__ scale) {
    const int row = blockIdx.x;
    const int t   = threadIdx.x;
    __shared__ float red[256];
    float m = 0.0f;
    #pragma unroll
    for (int i = 0; i < 16; ++i)
        m = fmaxf(m, fabsf(src[(size_t)row * D_IN + t + i * 256]));
    red[t] = m;
    __syncthreads();
    for (int s = 128; s > 0; s >>= 1) {
        if (t < s) red[t] = fmaxf(red[t], red[t + s]);
        __syncthreads();
    }
    if (t == 0) scale[row] = fmaxf(red[0], 1e-8f);
}

__global__ __launch_bounds__(256)
void k_quant_transpose_W(const float* __restrict__ W, const float* __restrict__ scb,
                         int8_t* __restrict__ CBt) {
    __shared__ __align__(16) unsigned char t[64 * 128];
    const int kt  = blockIdx.x;
    const int nb  = blockIdx.y;
    const int tid = threadIdx.x;
    #pragma unroll
    for (int i = 0; i < 8; ++i) {
        int task = tid + i * 256;
        int rr   = task >> 4;
        int c4   = (task & 15) * 4;
        const float4 v = *(const float4*)(W + (size_t)(nb * 128 + rr) * D_IN + kt * 64 + c4);
        float sc = 127.0f / scb[nb * 128 + rr];
        float q0 = fminf(fmaxf(rintf(v.x * sc), -127.0f), 127.0f);
        float q1 = fminf(fmaxf(rintf(v.y * sc), -127.0f), 127.0f);
        float q2 = fminf(fmaxf(rintf(v.z * sc), -127.0f), 127.0f);
        float q3 = fminf(fmaxf(rintf(v.w * sc), -127.0f), 127.0f);
        t[rr * 64 + c4 + 0] = (unsigned char)(int8_t)q0;
        t[rr * 64 + c4 + 1] = (unsigned char)(int8_t)q1;
        t[rr * 64 + c4 + 2] = (unsigned char)(int8_t)q2;
        t[rr * 64 + c4 + 3] = (unsigned char)(int8_t)q3;
    }
    __syncthreads();
    uint4* dst = (uint4*)(CBt + (size_t)nb * (D_IN * 128) + (size_t)kt * 8192);
    const uint4* srcv = (const uint4*)t;
    dst[tid]       = srcv[tid];
    dst[tid + 256] = srcv[tid + 256];
}

__global__ __launch_bounds__(256, 1)
void k_gemm_int8(const int8_t* __restrict__ CA, const int8_t* __restrict__ CBt,
                 const float* __restrict__ sca, const float* __restrict__ scb,
                 const float* __restrict__ x,  const float* __restrict__ W,
                 const float* __restrict__ bias,
                 const int* __restrict__ cnt, const int* __restrict__ list,
                 float* __restrict__ out) {
    __shared__ __align__(16) unsigned char smem[33280];

    const int tid  = threadIdx.x;
    const int lane = tid & 31;
    const int wave = tid >> 5;
    const int wt   = wave & 1;
    const int wf   = wave >> 1;
    const int tok0 = blockIdx.y * 128;
    const int fea0 = blockIdx.x * 128;

    const int arr   = (tid + 0)   >> 2, ac16  = (tid & 3) * 16;
    const int arr2  = (tid + 256) >> 2, ac162 = ((tid + 256) & 3) * 16;
    const uint4* btile = (const uint4*)(CBt + (size_t)blockIdx.x * (D_IN * 128));

    const v8i vzero = {0, 0, 0, 0, 0, 0, 0, 0};
    v8i acc[4][2];
    #pragma unroll
    for (int i = 0; i < 4; ++i)
        #pragma unroll
        for (int j = 0; j < 2; ++j) acc[i][j] = vzero;

    const int mrow = lane & 15;
    const int k0   = (lane >> 4) * 8;

    uint4 ra0, ra1, rb0, rb1;
    ra0 = *(const uint4*)(CA + (size_t)(tok0 + arr)  * D_IN + ac16);
    ra1 = *(const uint4*)(CA + (size_t)(tok0 + arr2) * D_IN + ac162);
    rb0 = btile[tid];
    rb1 = btile[tid + 256];
    *(uint4*)(smem + arr  * 64 + ac16)  = ra0;
    *(uint4*)(smem + arr2 * 64 + ac162) = ra1;
    ((uint4*)(smem + 8192))[tid]       = rb0;
    ((uint4*)(smem + 8192))[tid + 256] = rb1;
    __syncthreads();

    const int NK = D_IN / 64;
    for (int kt = 0; kt < NK; ++kt) {
        const int offA = (kt & 1) * 16384;
        const int offB = offA + 8192;
        if (kt + 1 < NK) {
            const int ko = (kt + 1) * 64;
            ra0 = *(const uint4*)(CA + (size_t)(tok0 + arr)  * D_IN + ko + ac16);
            ra1 = *(const uint4*)(CA + (size_t)(tok0 + arr2) * D_IN + ko + ac162);
            rb0 = btile[(kt + 1) * 512 + tid];
            rb1 = btile[(kt + 1) * 512 + tid + 256];
            if (kt + 2 < NK)
                __builtin_prefetch((const char*)&btile[(kt + 2) * 512 + tid], 0, 3);
        }

        v8i aF[4];
        #pragma unroll
        for (int i = 0; i < 4; ++i) {
            const unsigned char* ap = smem + offA + (wt * 64 + i * 16 + mrow) * 64;
            uint2 d0 = *(const uint2*)(ap + k0);
            uint2 d1 = *(const uint2*)(ap + k0 + 16);
            uint2 d2 = *(const uint2*)(ap + k0 + 32);
            uint2 d3 = *(const uint2*)(ap + k0 + 48);
            v8i a = {(int)d0.x, (int)d0.y, (int)d1.x, (int)d1.y,
                     (int)d2.x, (int)d2.y, (int)d3.x, (int)d3.y};
            aF[i] = a;
        }
        v8i bF[2];
        #pragma unroll
        for (int j = 0; j < 2; ++j) {
            int f0 = wf * 32 + j * 16;
            const unsigned char* bp = smem + offB + (f0 + mrow) * 64;
            uint2 e0 = *(const uint2*)(bp + k0);
            uint2 e1 = *(const uint2*)(bp + k0 + 16);
            uint2 e2 = *(const uint2*)(bp + k0 + 32);
            uint2 e3 = *(const uint2*)(bp + k0 + 48);
            v8i b = {(int)e0.x, (int)e0.y, (int)e1.x, (int)e1.y,
                     (int)e2.x, (int)e2.y, (int)e3.x, (int)e3.y};
            bF[j] = b;
        }
        #pragma unroll
        for (int i = 0; i < 4; ++i)
            #pragma unroll
            for (int j = 0; j < 2; ++j)
                acc[i][j] = __builtin_amdgcn_wmma_i32_16x16x64_iu8(
                    true, aF[i], true, bF[j], acc[i][j], false, false);

        if (kt + 1 < NK) {
            const int nA = 16384 - offA;
            const int nB = nA + 8192;
            *(uint4*)(smem + nA + arr  * 64 + ac16)  = ra0;
            *(uint4*)(smem + nA + arr2 * 64 + ac162) = ra1;
            ((uint4*)(smem + nB))[tid]       = rb0;
            ((uint4*)(smem + nB))[tid + 256] = rb1;
        }
        __syncthreads();
    }

    float (*xs)[NC]      = (float (*)[NC])(smem);
    float (*wsp)[NC + 1] = (float (*)[NC + 1])(smem + 16384);
    const int nOut = *cnt;
    const int cn   = nOut < NC ? nOut : NC;
    #pragma unroll
    for (int e = tid; e < 128 * NC; e += 256) {
        int rr = e >> 5;
        int ci = e & 31;
        float xv = 0.0f, wv = 0.0f;
        if (ci < cn) {
            int col = list[ci];
            xv = x[(size_t)(tok0 + rr) * D_IN + col];
            wv = W[(size_t)(fea0 + rr) * D_IN + col];
        }
        xs[rr][ci]  = xv;
        wsp[rr][ci] = wv;
    }
    __syncthreads();

    const float inv = 1.0f / (127.0f * 127.0f);
    const int   fl  = lane & 15;
    const int   rhi = (lane >> 4) * 8;
    #pragma unroll
    for (int i = 0; i < 4; ++i) {
        #pragma unroll
        for (int j = 0; j < 2; ++j) {
            const int flocal = wf * 32 + j * 16 + fl;
            const int fc     = fea0 + flocal;
            const float sb   = scb[fc];
            const float bb   = bias[fc];
            #pragma unroll
            for (int v = 0; v < 8; ++v) {
                const int rlocal = wt * 64 + i * 16 + v + rhi;
                const int tr     = tok0 + rlocal;
                float r = (float)acc[i][j][v] * (sca[tr] * sb) * inv + bb;
                float os = 0.0f;
                for (int ci = 0; ci < cn; ++ci)
                    os += xs[rlocal][ci] * wsp[flocal][ci];
                for (int ci = NC; ci < nOut; ++ci) {
                    int col = list[ci];
                    os += x[(size_t)tr * D_IN + col] * W[(size_t)fc * D_IN + col];
                }
                out[(size_t)tr * D_OUT + fc] = r + os;
            }
        }
    }
}

extern "C" void kernel_launch(void* const* d_in, const int* in_sizes, int n_in,
                              void* d_out, int out_size, void* d_ws, size_t ws_size,
                              hipStream_t stream) {
    const float* x    = (const float*)d_in[0];
    const float* W    = (const float*)d_in[1];
    const float* bias = (const float*)d_in[2];
    float* out = (float*)d_out;

    char* ws = (char*)d_ws;
    unsigned int* colmax = (unsigned int*)(ws + 0);
    int*    cnt  = (int*)  (ws + 16384);
    int*    list = (int*)  (ws + 16512);
    float*  sca  = (float*)(ws + 33024);
    float*  scb  = (float*)(ws + 49408);
    int8_t* CA   = (int8_t*)(ws + 131072);
    int8_t* CBt  = (int8_t*)(ws + 131072 + (size_t)NTOK * D_IN);

    k_init<<<16, 256, 0, stream>>>(colmax, cnt);
    k_colmax<<<dim3(D_IN / 256, 64), 256, 0, stream>>>(x, colmax);
    k_build_list<<<1, 1, 0, stream>>>(colmax, cnt, list);
    k_quant_x<<<NTOK, 256, 0, stream>>>(x, CA, sca, colmax);
    k_rowabsmax_W<<<D_OUT, 256, 0, stream>>>(W, scb);
    k_quant_transpose_W<<<dim3(D_IN / 64, D_OUT / 128), 256, 0, stream>>>(W, scb, CBt);
    k_gemm_int8<<<dim3(D_OUT / 128, NTOK / 128), 256, 0, stream>>>(
        CA, CBt, sca, scb, x, W, bias, cnt, list, out);
}
